// AttentionBase_8040178778698
// MI455X (gfx1250) — matrix operand order repaired and checked by reading, not compiled, not run on hardware
//
#include <hip/hip_runtime.h>

#define SEQ   4096
#define MID   512
#define NH    8
#define HD    64
#define BATCH 2

typedef __attribute__((ext_vector_type(16))) __bf16         v16bf;
typedef __attribute__((ext_vector_type(16))) unsigned short v16us;
typedef __attribute__((ext_vector_type(8)))  float          v8f;
typedef __attribute__((ext_vector_type(8)))  unsigned       v8u;
typedef __attribute__((ext_vector_type(4)))  unsigned       v4u;
typedef __attribute__((ext_vector_type(4)))  unsigned       u32x4;
typedef __attribute__((ext_vector_type(8)))  int            i32x8;
typedef __attribute__((ext_vector_type(4)))  int            i32x4;

static __device__ __forceinline__ unsigned short f2bf(float f) {
    unsigned u = __builtin_bit_cast(unsigned, f);
    u += 0x7FFFu + ((u >> 16) & 1u);          // round-to-nearest-even
    return (unsigned short)(u >> 16);
}

static __device__ __forceinline__ v8f wmma_bf16(v16us a, v16us b, v8f c) {
    return __builtin_amdgcn_wmma_f32_16x16x32_bf16(
        false, __builtin_bit_cast(v16bf, a),
        false, __builtin_bit_cast(v16bf, b),
        (short)0, c, false, false);
}

// ---------------------------------------------------------------------------
// TDM 2D tile load: D# per CDNA5 ISA 8.3/8.4 (count=1, data_size=2B, type=2)
// 6-arg builtin variant: (g0, g1, g2, g3, g_extra, cpol)
// ---------------------------------------------------------------------------
static __device__ __forceinline__ void tdm_load_2d(
    unsigned lds_off, const void* gaddr,
    unsigned tensor_d0, unsigned tensor_d1, unsigned d0_stride,
    unsigned tile_d0, unsigned tile_d1)
{
    unsigned long long ga = (unsigned long long)gaddr;
    u32x4 g0;
    g0[0] = 1u;                                               // count=1, user mode
    g0[1] = lds_off;                                          // LDS byte address
    g0[2] = (unsigned)(ga & 0xFFFFFFFFu);                     // global_addr[31:0]
    g0[3] = (unsigned)((ga >> 32) & 0x01FFFFFFu) | 0x80000000u; // addr[56:32] | type=2
    i32x8 g1;
    g1[0] = (int)(1u << 16);                                  // data_size = 1 (2 bytes)
    g1[1] = (int)((tensor_d0 & 0xFFFFu) << 16);               // tensor_dim0 lo16
    g1[2] = (int)(((tensor_d0 >> 16) & 0xFFFFu) | ((tensor_d1 & 0xFFFFu) << 16));
    g1[3] = (int)(((tensor_d1 >> 16) & 0xFFFFu) | ((tile_d0 & 0xFFFFu) << 16));
    g1[4] = (int)(tile_d1 & 0xFFFFu);                         // tile_dim2 = 0
    g1[5] = (int)d0_stride;                                   // dim0_stride[31:0]
    g1[6] = 0;
    g1[7] = 0;
    i32x4 z4 = {0, 0, 0, 0};
    i32x8 z8 = {0, 0, 0, 0, 0, 0, 0, 0};
    __builtin_amdgcn_tensor_load_to_lds(g0, g1, z4, z4, z8, 0);
}

// ---------------------------------------------------------------------------
// W_out [512,512] fp32 -> Wt[n][k] bf16 (transposed: B-fragments read rows)
// ---------------------------------------------------------------------------
__global__ void __launch_bounds__(256)
convert_w_kernel(const float* __restrict__ W, unsigned short* __restrict__ Wt) {
    int idx = blockIdx.x * 256 + threadIdx.x;
    int k = idx >> 9, n = idx & 511;
    Wt[(long)n * 512 + k] = f2bf(W[(long)k * 512 + n]);
}

// ---------------------------------------------------------------------------
// K,V fp32 [b,n,512] -> Kb bf16 [b,h,key,dim], Vt bf16 [b,h,dim,key]
// ---------------------------------------------------------------------------
__global__ void __launch_bounds__(256)
convert_kv_kernel(const float* __restrict__ K, const float* __restrict__ V,
                  unsigned short* __restrict__ Kb, unsigned short* __restrict__ Vt) {
    long idx = (long)blockIdx.x * 256 + threadIdx.x;       // 2*4096*512 total
    int  b = (int)(idx >> 21);
    int  key = (int)((idx >> 9) & 4095);
    int  c = (int)(idx & 511);
    int  h = c >> 6, d = c & 63;
    int  bh = b * NH + h;
    unsigned short kv = f2bf(K[idx]);
    unsigned short vv = f2bf(V[idx]);
    Kb[((long)bh * SEQ + key) * HD + d] = kv;
    Vt[((long)bh * HD + d) * SEQ + key] = vv;
}

// ---------------------------------------------------------------------------
// Brel[h][rel+4095] = table[bucket(rel)][h] * SCALE  (T5 bucket, logf here once)
// ---------------------------------------------------------------------------
__global__ void __launch_bounds__(256)
bias_kernel(const float* __restrict__ table, float* __restrict__ Brel) {
    int idx = blockIdx.x * 256 + threadIdx.x;
    if (idx >= NH * 8191) return;
    int h = idx / 8191;
    int rel = (idx % 8191) - 4095;
    int ret = (rel >= 0) ? 32 : 0;
    int na  = rel < 0 ? -rel : rel;
    int vl;
    if (na < 16) vl = na;
    else {
        int vv = 16 + (int)(__logf((float)na * 0.0625f) * 0.4808983469629878f * 16.0f);
        vl = vv < 31 ? vv : 31;
    }
    Brel[idx] = table[(ret + vl) * NH + h] * 0.125f;
}

// ---------------------------------------------------------------------------
// Flash attention: (batch, head, 64 q-rows) per block, 4 waves.
// TDM double-buffers K/V tiles HBM(L2) -> LDS; WMMA bf16 for QK^T and PV.
// ---------------------------------------------------------------------------
__global__ void __launch_bounds__(128)
attn_kernel(const float* __restrict__ Q,
            const unsigned short* __restrict__ Kb,
            const unsigned short* __restrict__ Vt,
            const float* __restrict__ Brel,
            unsigned short* __restrict__ X)
{
    __shared__ __align__(16) unsigned short Kl[2][64 * 64];
    __shared__ __align__(16) unsigned short Vl[2][64 * 64];
    __shared__ __align__(16) unsigned short Pl[4][16 * 64];
    __shared__ float Bw[128];

    const int tid  = threadIdx.x;
    const int lane = tid & 31;
    const int wave = tid >> 5;
    const int hlf  = (lane >> 4) & 1;
    const int l16  = lane & 15;

    const int qtile = blockIdx.x & 63;
    const int bh    = blockIdx.x >> 6;
    const int head  = bh & 7;
    const int batch = bh >> 3;
    const int q0    = qtile * 64;

    const unsigned short* Kh = Kb + (long)bh * SEQ * HD;   // [key][dim], stride 64
    const unsigned short* Vh = Vt + (long)bh * HD * SEQ;   // [dim][key], stride 4096
    const float*          Bh = Brel + head * 8191;

    // --- Q A-fragments (fp32 global -> bf16, once)
    const int  qrow  = q0 + wave * 16 + l16;
    const long qbase = ((long)(batch * SEQ + qrow)) * MID + head * HD;
    v16us aq[2];
#pragma unroll
    for (int p = 0; p < 2; ++p)
#pragma unroll
        for (int e = 0; e < 16; ++e) {
            int kd = p * 32 + hlf * 16 + e;
            aq[p][e] = f2bf(Q[qbase + kd]);
        }

    v8f   o[4];
    float m[8], lsum[8];
#pragma unroll
    for (int t = 0; t < 4; ++t)
#pragma unroll
        for (int i = 0; i < 8; ++i) o[t][i] = 0.0f;
#pragma unroll
    for (int i = 0; i < 8; ++i) { m[i] = -1e30f; lsum[i] = 0.0f; }

    const unsigned kl_off0 = (unsigned)(size_t)&Kl[0][0];
    const unsigned kl_off1 = (unsigned)(size_t)&Kl[1][0];
    const unsigned vl_off0 = (unsigned)(size_t)&Vl[0][0];
    const unsigned vl_off1 = (unsigned)(size_t)&Vl[1][0];

    // --- prologue: wave0 issues TDM for tile 0
    if (wave == 0) {
        tdm_load_2d(kl_off0, Kh + (long)0 * HD, 64, 64, 64, 64, 64);
        tdm_load_2d(vl_off0, Vh + 0,            64, 64, SEQ, 64, 64);
    }

    for (int it = 0; it < SEQ / 64; ++it) {
        const int kb  = it * 64;
        const int buf = it & 1;

        // stage the 127-entry bias window for this key tile (bias pre-scaled)
        {
            int j = tid;                        // 0..127 (entry 127 unused)
            int rel = kb - q0 - 63 + j;         // rel in [-4095, 4095]
            Bw[j] = Bh[rel + 4095];
        }

        // prefetch next tile into the other buffer; wait for this tile
        if (wave == 0) {
            if (it + 1 < SEQ / 64) {
                tdm_load_2d(buf ? kl_off0 : kl_off1, Kh + (long)(kb + 64) * HD,
                            64, 64, 64, 64, 64);
                tdm_load_2d(buf ? vl_off0 : vl_off1, Vh + (kb + 64),
                            64, 64, SEQ, 64, 64);
                __builtin_amdgcn_s_wait_tensorcnt((short)2);
            } else {
                __builtin_amdgcn_s_wait_tensorcnt((short)0);
            }
        }
        __syncthreads();   // tile `it` + bias window visible to all waves

        // --- S = Q K^T : 4 key n-tiles, 2 chained WMMAs each (K-dim 64)
        v8f s[4];
#pragma unroll
        for (int t = 0; t < 4; ++t) {
            v8f c;
#pragma unroll
            for (int i = 0; i < 8; ++i) c[i] = 0.0f;
#pragma unroll
            for (int p = 0; p < 2; ++p) {
                v16us bk = __builtin_bit_cast(
                    v16us, *(const v8u*)&Kl[buf][(t * 16 + l16) * 64 + p * 32 + hlf * 16]);
                c = wmma_bf16(aq[p], bk, c);
            }
            s[t] = c;
        }

        // --- bias + scale + online softmax (C-layout: row = hlf*8+i, col = l16)
#pragma unroll
        for (int i = 0; i < 8; ++i) {
            const int dq = wave * 16 + hlf * 8 + i;      // qr - q0
            float mx = -1e30f;
#pragma unroll
            for (int t = 0; t < 4; ++t) {
                float bias = Bw[63 + t * 16 + l16 - dq];
                float sv   = fmaf(s[t][i], 0.125f, bias);
                s[t][i] = sv;
                mx = fmaxf(mx, sv);
            }
#pragma unroll
            for (int msk = 1; msk < 16; msk <<= 1)
                mx = fmaxf(mx, __shfl_xor(mx, msk, 32));
            float newm  = fmaxf(m[i], mx);
            float alpha = __expf(m[i] - newm);
            float rs    = 0.0f;
#pragma unroll
            for (int t = 0; t < 4; ++t) {
                float p = __expf(s[t][i] - newm);
                s[t][i] = p;
                rs += p;
            }
#pragma unroll
            for (int msk = 1; msk < 16; msk <<= 1)
                rs += __shfl_xor(rs, msk, 32);
            lsum[i] = lsum[i] * alpha + rs;
            m[i]    = newm;
#pragma unroll
            for (int t = 0; t < 4; ++t) o[t][i] *= alpha;
        }

        // --- P: C-layout -> A-layout via per-wave LDS (same-wave DS is in-order)
#pragma unroll
        for (int t = 0; t < 4; ++t)
#pragma unroll
            for (int i = 0; i < 8; ++i)
                Pl[wave][(hlf * 8 + i) * 64 + t * 16 + l16] = f2bf(s[t][i]);

        v16us ap[2];
        const unsigned short* prow = &Pl[wave][l16 * 64];
#pragma unroll
        for (int p = 0; p < 2; ++p) {
            v4u lo = *(const v4u*)(prow + p * 32 + hlf * 16);
            v4u hi = *(const v4u*)(prow + p * 32 + hlf * 16 + 8);
            ap[p] = __builtin_bit_cast(v16us,
                        __builtin_shufflevector(lo, hi, 0, 1, 2, 3, 4, 5, 6, 7));
        }

        // --- O += P V : V tile is dim-major so B-fragments are contiguous
#pragma unroll
        for (int t = 0; t < 4; ++t) {
            const int dim = t * 16 + l16;
#pragma unroll
            for (int p = 0; p < 2; ++p) {
                v16us bv = __builtin_bit_cast(
                    v16us, *(const v8u*)&Vl[buf][dim * 64 + p * 32 + hlf * 16]);
                o[t] = wmma_bf16(ap[p], bv, o[t]);
            }
        }
        __syncthreads();   // all reads of buf done before TDM overwrites it
    }

    // --- normalize, write X bf16 [b, n, h*d]
#pragma unroll
    for (int t = 0; t < 4; ++t)
#pragma unroll
        for (int i = 0; i < 8; ++i) {
            int qr  = q0 + wave * 16 + hlf * 8 + i;
            int dim = t * 16 + l16;
            X[((long)(batch * SEQ + qr)) * MID + head * HD + dim] =
                f2bf(o[t][i] / lsum[i]);
        }
}

// ---------------------------------------------------------------------------
// out = X(bf16) @ W + b : 64x64 tile/block, 16 chained K=32 WMMAs per tile
// ---------------------------------------------------------------------------
__global__ void __launch_bounds__(128)
proj_kernel(const unsigned short* __restrict__ X, const unsigned short* __restrict__ Wt,
            const float* __restrict__ bias, float* __restrict__ out)
{
    const int tid  = threadIdx.x;
    const int lane = tid & 31;
    const int wave = tid >> 5;
    const int hlf  = (lane >> 4) & 1;
    const int l16  = lane & 15;

    const int row0 = (blockIdx.x >> 3) * 64 + wave * 16;
    const int n0   = (blockIdx.x & 7) * 64;

    v8f acc[4];
#pragma unroll
    for (int t = 0; t < 4; ++t)
#pragma unroll
        for (int i = 0; i < 8; ++i) acc[t][i] = 0.0f;

    const unsigned short* xrow = X + (long)(row0 + l16) * 512;

    for (int kb = 0; kb < 512; kb += 32) {
        v4u lo = *(const v4u*)(xrow + kb + hlf * 16);
        v4u hi = *(const v4u*)(xrow + kb + hlf * 16 + 8);
        v16us ax = __builtin_bit_cast(v16us,
                       __builtin_shufflevector(lo, hi, 0, 1, 2, 3, 4, 5, 6, 7));
#pragma unroll
        for (int t = 0; t < 4; ++t) {
            int n = n0 + t * 16 + l16;
            v16us bw = __builtin_bit_cast(
                v16us, *(const v8u*)(Wt + (long)n * 512 + kb + hlf * 16));
            acc[t] = wmma_bf16(ax, bw, acc[t]);
        }
    }

#pragma unroll
    for (int t = 0; t < 4; ++t) {
        int   n  = n0 + t * 16 + l16;
        float bn = bias[n];
#pragma unroll
        for (int i = 0; i < 8; ++i)
            out[(long)(row0 + hlf * 8 + i) * 512 + n] = acc[t][i] + bn;
    }
}

// ---------------------------------------------------------------------------
extern "C" void kernel_launch(void* const* d_in, const int* in_sizes, int n_in,
                              void* d_out, int out_size, void* d_ws, size_t ws_size,
                              hipStream_t stream)
{
    (void)in_sizes; (void)n_in; (void)out_size; (void)ws_size;
    const float* q     = (const float*)d_in[0];
    const float* k     = (const float*)d_in[1];
    const float* v     = (const float*)d_in[2];
    const float* table = (const float*)d_in[3];
    const float* W     = (const float*)d_in[4];
    const float* b     = (const float*)d_in[5];
    float*       out   = (float*)d_out;

    unsigned short* X    = (unsigned short*)d_ws;                    // 8 MB bf16
    unsigned short* Wt   = X  + (size_t)BATCH * SEQ * MID;           // 512 KB
    unsigned short* Kb   = Wt + (size_t)MID * MID;                   // 8 MB
    unsigned short* Vt   = Kb + (size_t)BATCH * SEQ * MID;           // 8 MB
    float*          Brel = (float*)(Vt + (size_t)BATCH * SEQ * MID); // 256 KB

    hipLaunchKernelGGL(convert_w_kernel,  dim3(1024),  dim3(256), 0, stream, W, Wt);
    hipLaunchKernelGGL(convert_kv_kernel, dim3(16384), dim3(256), 0, stream, k, v, Kb, Vt);
    hipLaunchKernelGGL(bias_kernel,       dim3(256),   dim3(256), 0, stream, table, Brel);
    hipLaunchKernelGGL(attn_kernel,       dim3(1024),  dim3(128), 0, stream, q, Kb, Vt, Brel, X);
    hipLaunchKernelGGL(proj_kernel,       dim3(1024),  dim3(128), 0, stream, X, Wt, b, out);
}
